// MambaBlock_21895743275797
// MI455X (gfx1250) — matrix operand order repaired and checked by reading, not compiled, not run on hardware
//
#include <hip/hip_runtime.h>
#include <hip/hip_bf16.h>

// ---------------------------------------------------------------------------
// Mamba block forward for MI455X (gfx1250), wave32, WMMA bf16 matrix cores.
// ---------------------------------------------------------------------------

#define D_MODEL 1024
#define D_INNER 2048
#define D_STATE 16
#define D_CONV  4
#define DT_RANK 64
#define BATCH   2
#define SEQ     2048
#define ML      (BATCH * SEQ)          // 4096 rows for all GEMMs

typedef __attribute__((ext_vector_type(16))) __bf16 v16bf;
typedef __attribute__((ext_vector_type(8)))  float  v8f;

union FragAB { v16bf v; unsigned int u[8]; };

__device__ __forceinline__ unsigned int pack_bf16x2(float x, float y) {
  // native casts -> backend can use v_cvt_pk_bf16_f32
  union { __bf16 b[2]; unsigned int u; } p;
  p.b[0] = (__bf16)x; p.b[1] = (__bf16)y;
  return p.u;
}

__device__ __forceinline__ float softplusf(float x) {
  return (x > 20.f) ? x : log1pf(__expf(x));
}
__device__ __forceinline__ float siluf(float x) {
  return x * (1.f / (1.f + __expf(-x)));
}

// ---------------------------------------------------------------------------
// Generic GEMM:  Out[M,N] = act( X[M,K(lda)] * W[N,K(ldw)]^T  (+ bias[N]) )
// Block: 256 threads = 8 waves; block tile 128(M) x 64(N); K-step 32.
// Wave w computes a 32x32 tile at (wm=(w>>1)*32, wn=(w&1)*32): 4 WMMAs/step.
// Double-buffered LDS staging (f32 -> packed bf16 u32, pitch 17 dwords).
// act: 0 = none, 1 = softplus(x + bias).  M % 128 == 0; N ragged (guarded).
// ---------------------------------------------------------------------------
__global__ __launch_bounds__(256)
void gemm_bf16_wmma(const float* __restrict__ X, const float* __restrict__ W,
                    float* __restrict__ Out, const float* __restrict__ bias,
                    int M, int N, int K, int lda, int ldw, int ldc, int act)
{
  __shared__ unsigned int Asl[2][128][17];  // 128 x 32 bf16 (16 u32 + 1 pad)
  __shared__ unsigned int Bsl[2][64][17];   // 64(N) x 32(K) bf16

  const int tid  = threadIdx.x;
  const int lane = tid & 31;
  const int w    = tid >> 5;
  const int m0   = blockIdx.y * 128;
  const int n0   = blockIdx.x * 64;
  const int wm   = (w >> 1) * 32;
  const int wn   = (w & 1) * 32;
  const int half = lane >> 4;     // lane group selects K sub-range
  const int mr   = lane & 15;     // A row / B column within 16-wide tile

  v8f acc00 = {}, acc01 = {}, acc10 = {}, acc11 = {};

  auto stage = [&](int buf, int kk) {
    // A tile: 128x32 f32 = 1024 float4 -> 4 per thread
#pragma unroll
    for (int i = 0; i < 4; ++i) {
      int f4 = tid + 256 * i;
      int r = f4 >> 3, c4 = f4 & 7;
      const float4 v = *(const float4*)(X + (size_t)(m0 + r) * lda + kk + c4 * 4);
      Asl[buf][r][c4 * 2]     = pack_bf16x2(v.x, v.y);
      Asl[buf][r][c4 * 2 + 1] = pack_bf16x2(v.z, v.w);
    }
    // B tile: 64x32 f32 = 512 float4 -> 2 per thread (guard ragged N)
#pragma unroll
    for (int i = 0; i < 2; ++i) {
      int f4 = tid + 256 * i;
      int r = f4 >> 3, c4 = f4 & 7;
      float4 v = make_float4(0.f, 0.f, 0.f, 0.f);
      if (n0 + r < N)
        v = *(const float4*)(W + (size_t)(n0 + r) * ldw + kk + c4 * 4);
      Bsl[buf][r][c4 * 2]     = pack_bf16x2(v.x, v.y);
      Bsl[buf][r][c4 * 2 + 1] = pack_bf16x2(v.z, v.w);
    }
  };

  stage(0, 0);
  __syncthreads();

  for (int kk = 0; kk < K; kk += 32) {
    const int cur = (kk >> 5) & 1;
    if (kk + 32 < K) stage(cur ^ 1, kk + 32);   // overlap with compute

    // ---- fragments per ISA 7.12.2 (u32 index = K/2) -----------------------
    // A 16x32 bf16: lane(0-15)=row M; pair q: K/2 = q+4h (q<4), q+4+4h (q>=4)
    FragAB a0, a1, b0, b1;
#pragma unroll
    for (int q = 0; q < 8; ++q) {
      int kw = (q < 4) ? (q + 4 * half) : (q + 4 + 4 * half);
      a0.u[q] = Asl[cur][wm + mr][kw];
      a1.u[q] = Asl[cur][wm + 16 + mr][kw];
    }
    // B 32x16 bf16: lane(0-15)=col N; pair q: K/2 = q+4h (q<4), q+4+4h (q>=4)
#pragma unroll
    for (int q = 0; q < 8; ++q) {
      int kw = (q < 4) ? (q + 4 * half) : (q + 4 + 4 * half);
      b0.u[q] = Bsl[cur][wn + mr][kw];
      b1.u[q] = Bsl[cur][wn + 16 + mr][kw];
    }

    acc00 = __builtin_amdgcn_wmma_f32_16x16x32_bf16(false, a0.v, false, b0.v,
                                                    (short)0, acc00, false, false);
    acc01 = __builtin_amdgcn_wmma_f32_16x16x32_bf16(false, a0.v, false, b1.v,
                                                    (short)0, acc01, false, false);
    acc10 = __builtin_amdgcn_wmma_f32_16x16x32_bf16(false, a1.v, false, b0.v,
                                                    (short)0, acc10, false, false);
    acc11 = __builtin_amdgcn_wmma_f32_16x16x32_bf16(false, a1.v, false, b1.v,
                                                    (short)0, acc11, false, false);
    __syncthreads();
  }

  // ---- epilogue: C/D layout VGPR r -> M = r + 8*half, col = lane%16 --------
  const int col0 = n0 + wn + mr;
  const int col1 = col0 + 16;
#pragma unroll
  for (int r = 0; r < 8; ++r) {
    int row0 = m0 + wm + r + 8 * half;
    int row1 = row0 + 16;
    float v00 = acc00[r], v01 = acc01[r], v10 = acc10[r], v11 = acc11[r];
    if (act == 1) {
      if (col0 < N) { v00 = softplusf(v00 + bias[col0]);
                      v10 = softplusf(v10 + bias[col0]); }
      if (col1 < N) { v01 = softplusf(v01 + bias[col1]);
                      v11 = softplusf(v11 + bias[col1]); }
    }
    if (col0 < N) { Out[(size_t)row0 * ldc + col0] = v00;
                    Out[(size_t)row1 * ldc + col0] = v10; }
    if (col1 < N) { Out[(size_t)row0 * ldc + col1] = v01;
                    Out[(size_t)row1 * ldc + col1] = v11; }
  }
}

// ---------------------------------------------------------------------------
// Depthwise causal conv (k=4) + bias + SiLU.
// xz row layout: [b,l, 0:2048]=xc, [2048:4096]=z.  One thread per (b,l,d).
// ---------------------------------------------------------------------------
__global__ __launch_bounds__(256)
void conv_silu_kernel(const float* __restrict__ xz, const float* __restrict__ cw,
                      const float* __restrict__ cb, float* __restrict__ u)
{
  size_t idx = (size_t)blockIdx.x * 256 + threadIdx.x;
  if (idx >= (size_t)BATCH * SEQ * D_INNER) return;
  int d = (int)(idx & (D_INNER - 1));
  int l = (int)((idx >> 11) & (SEQ - 1));
  int b = (int)(idx >> 22);
  float acc = cb[d];
#pragma unroll
  for (int j = 0; j < D_CONV; ++j) {
    int ll = l - (D_CONV - 1) + j;
    if (ll >= 0)
      acc += xz[((size_t)(b * SEQ + ll)) * (2 * D_INNER) + d] * cw[d * D_CONV + j];
  }
  u[idx] = siluf(acc);
}

// ---------------------------------------------------------------------------
// Selective scan. One channel (b,d) per lane; h[16], A[16] in registers.
// B/C (shared per timestep) staged through LDS in 64-step chunks.
// Writes yact = (y + u*D) * silu(z) in-place over the delta buffer.
// grid = (D_INNER/256, BATCH), block = 256.
// ---------------------------------------------------------------------------
#define SCAN_CHUNK 64
__global__ __launch_bounds__(256)
void scan_kernel(const float* __restrict__ xdbl,   // [ML,96] (B at 64, C at 80)
                 float* __restrict__ delta_yact,   // [ML,D_INNER] in, yact out
                 const float* __restrict__ u,      // [ML,D_INNER]
                 const float* __restrict__ xz,     // [ML,2*D_INNER] (z half)
                 const float* __restrict__ A_log,  // [D_INNER,16]
                 const float* __restrict__ Dvec)   // [D_INNER]
{
  __shared__ float BCs[SCAN_CHUNK * 32];  // per step: 16 B then 16 C
  const int tid = threadIdx.x;
  const int b   = blockIdx.y;
  const int d   = blockIdx.x * 256 + tid;

  float A[D_STATE], h[D_STATE];
#pragma unroll
  for (int n = 0; n < D_STATE; ++n) {
    A[n] = -__expf(A_log[(size_t)d * D_STATE + n]);
    h[n] = 0.f;
  }
  const float Dd = Dvec[d];

  for (int l0 = 0; l0 < SEQ; l0 += SCAN_CHUNK) {
    // cooperative load of B/C for this chunk: 64 steps * 32 floats
#pragma unroll
    for (int i = 0; i < (SCAN_CHUNK * 32) / 256; ++i) {
      int f = tid + 256 * i;
      int step = f >> 5, off = f & 31;
      BCs[f] = xdbl[((size_t)(b * SEQ + l0 + step)) * 96 + DT_RANK + off];
    }
    __syncthreads();

    for (int t = 0; t < SCAN_CHUNK; ++t) {
      const size_t row = (size_t)(b * SEQ + l0 + t);
      const size_t idx = row * D_INNER + d;
      const float dt = delta_yact[idx];
      const float uu = u[idx];
      const float zz = xz[row * (2 * D_INNER) + D_INNER + d];
      const float du = dt * uu;
      const float* Bsh = &BCs[t * 32];
      const float* Csh = &BCs[t * 32 + 16];
      float y = 0.f;
#pragma unroll
      for (int n = 0; n < D_STATE; ++n) {
        float dA = __expf(dt * A[n]);
        h[n] = fmaf(dA, h[n], du * Bsh[n]);
        y = fmaf(h[n], Csh[n], y);
      }
      delta_yact[idx] = (y + uu * Dd) * siluf(zz);  // same slot this thread read
    }
    __syncthreads();
  }
}

// ---------------------------------------------------------------------------
// Launch
// ---------------------------------------------------------------------------
extern "C" void kernel_launch(void* const* d_in, const int* in_sizes, int n_in,
                              void* d_out, int out_size, void* d_ws, size_t ws_size,
                              hipStream_t stream)
{
  const float* x      = (const float*)d_in[0];  // [ML, 1024]
  const float* w_in   = (const float*)d_in[1];  // [4096, 1024]
  const float* conv_w = (const float*)d_in[2];  // [2048, 1, 4]
  const float* conv_b = (const float*)d_in[3];  // [2048]
  const float* w_x    = (const float*)d_in[4];  // [96, 2048]
  const float* w_dt   = (const float*)d_in[5];  // [2048, 64]
  const float* b_dt   = (const float*)d_in[6];  // [2048]
  const float* A_log  = (const float*)d_in[7];  // [2048, 16]
  const float* Dvec   = (const float*)d_in[8];  // [2048]
  const float* w_out  = (const float*)d_in[9];  // [1024, 2048]
  float* out = (float*)d_out;                   // [ML, 1024]

  float* ws_xz    = (float*)d_ws;               // [ML, 4096]   67.1 MB
  float* ws_u     = ws_xz  + (size_t)ML * 2 * D_INNER;   // [ML, 2048] 33.5 MB
  float* ws_xdbl  = ws_u   + (size_t)ML * D_INNER;       // [ML, 96]    1.6 MB
  float* ws_delta = ws_xdbl + (size_t)ML * 96;           // [ML, 2048] 33.5 MB (reused as yact)

  // 1) xz = x @ w_in^T            (M=4096, N=4096, K=1024)
  gemm_bf16_wmma<<<dim3(4096 / 64, ML / 128), 256, 0, stream>>>(
      x, w_in, ws_xz, nullptr, ML, 2 * D_INNER, D_MODEL,
      D_MODEL, D_MODEL, 2 * D_INNER, 0);

  // 2) u = silu(depthwise_conv(xc) + conv_b)
  conv_silu_kernel<<<((size_t)ML * D_INNER) / 256, 256, 0, stream>>>(
      ws_xz, conv_w, conv_b, ws_u);

  // 3) x_dbl = u @ w_x^T          (M=4096, N=96, K=2048)
  gemm_bf16_wmma<<<dim3(2, ML / 128), 256, 0, stream>>>(
      ws_u, w_x, ws_xdbl, nullptr, ML, DT_RANK + 2 * D_STATE, D_INNER,
      D_INNER, D_INNER, 96, 0);

  // 4) delta = softplus(x_dbl[:, :64] @ w_dt^T + b_dt)  (M=4096, N=2048, K=64)
  gemm_bf16_wmma<<<dim3(D_INNER / 64, ML / 128), 256, 0, stream>>>(
      ws_xdbl, w_dt, ws_delta, b_dt, ML, D_INNER, DT_RANK,
      96, DT_RANK, D_INNER, 1);

  // 5) selective scan + output gating (in-place yact over delta buffer)
  scan_kernel<<<dim3(D_INNER / 256, BATCH), 256, 0, stream>>>(
      ws_xdbl, ws_delta, ws_u, ws_xz, A_log, Dvec);

  // 6) out = yact @ w_out^T       (M=4096, N=1024, K=2048)
  gemm_bf16_wmma<<<dim3(D_MODEL / 64, ML / 128), 256, 0, stream>>>(
      ws_delta, w_out, out, nullptr, ML, D_MODEL, D_INNER,
      D_INNER, D_INNER, D_MODEL, 0);
}
